// PointNet2EncoderModel_old_8710193676884
// MI455X (gfx1250) — matrix operand order repaired and checked by reading, not compiled, not run on hardware
//
#include <hip/hip_runtime.h>
#include <hip/hip_bf16.h>
#include <math.h>

#define BT 16            // B_*T_ = 8*2 point clouds
#define BN_INV 0.9999950000374997f   // 1/sqrt(1+1e-5)

typedef __bf16 bf16x16 __attribute__((ext_vector_type(16)));
typedef float  f32x8   __attribute__((ext_vector_type(8)));

__device__ inline float apply_act(float x, int act)
{
  if (act == 1) return fmaxf(x, 0.f);
  if (act == 2) return 0.5f * x * (1.f + erff(x * 0.70710678118654752f));
  return x;
}

__device__ inline void fill8(bf16x16& a, int base, float4 u, float4 v)
{
  a[base+0] = (__bf16)u.x; a[base+1] = (__bf16)u.y;
  a[base+2] = (__bf16)u.z; a[base+3] = (__bf16)u.w;
  a[base+4] = (__bf16)v.x; a[base+5] = (__bf16)v.y;
  a[base+6] = (__bf16)v.z; a[base+7] = (__bf16)v.w;
}

__device__ inline void load_a_frag(bf16x16& a, const float* Xr, int k0, int h)
{
  const float4* p0 = reinterpret_cast<const float4*>(Xr + k0 + 16*h);
  fill8(a, 0, p0[0], p0[1]);
  fill8(a, 8, p0[2], p0[3]);
}

// ============================================================================
// GEMM with pre-staged bf16 fragment-layout weights.
// Wf layout: fragment for (kc, nt) tile, lane l, elements e=0..15 at
//   Wf[((kc*(N/16) + nt)*32 + l)*16 + e],  e=2v+j <-> k=kc*32+2v+16*(l>>4)+j,
//   n = nt*16 + (l&15).   One b128-pair load per B fragment.
// Y[M,N] = act( (X[M,K] @ W + bias)*(g*BN_INV) + beta ).  K % 32 == 0.
// Wave computes 32x32 tile; block = 4 waves over 128 columns.
// ============================================================================
__global__ void gemm_fw(const float* __restrict__ X, const __bf16* __restrict__ Wf,
                        const float* __restrict__ bias, const float* __restrict__ g,
                        const float* __restrict__ beta, float* __restrict__ Y,
                        int M, int K, int N, int act)
{
  const int lane = threadIdx.x & 31;
  const int wave = threadIdx.x >> 5;
  const int m0   = blockIdx.x * 32;
  const int n0   = blockIdx.y * 128 + wave * 32;
  if (n0 >= N) return;
  const int h  = lane >> 4;
  const int mn = lane & 15;
  const int ntiles = N >> 4;
  const int nt0 = n0 >> 4;

  const float* Xr0 = X + (size_t)(m0 + mn) * K;
  const float* Xr1 = X + (size_t)(m0 + 16 + mn) * K;
  const int nc0 = n0 + mn, nc1 = n0 + 16 + mn;

  f32x8 c00 = {0.f,0.f,0.f,0.f,0.f,0.f,0.f,0.f};
  f32x8 c01 = c00, c10 = c00, c11 = c00;

  const int KC = K >> 5;
  for (int kc = 0; kc < KC; ++kc) {
    const int k0 = kc << 5;
    bf16x16 a0, a1;
    load_a_frag(a0, Xr0, k0, h);
    load_a_frag(a1, Xr1, k0, h);
    const __bf16* wp = Wf + (((size_t)kc * ntiles + nt0) * 32 + lane) * 16;
    bf16x16 b0 = *reinterpret_cast<const bf16x16*>(wp);
    bf16x16 b1 = *reinterpret_cast<const bf16x16*>(wp + 32 * 16);

    __builtin_prefetch(Xr0 + k0 + 32, 0, 3);
    __builtin_prefetch(wp + (size_t)ntiles * 32 * 16, 0, 3);

    c00 = __builtin_amdgcn_wmma_f32_16x16x32_bf16(false, a0, false, b0, (short)0, c00, false, false);
    c01 = __builtin_amdgcn_wmma_f32_16x16x32_bf16(false, a0, false, b1, (short)0, c01, false, false);
    c10 = __builtin_amdgcn_wmma_f32_16x16x32_bf16(false, a1, false, b0, (short)0, c10, false, false);
    c11 = __builtin_amdgcn_wmma_f32_16x16x32_bf16(false, a1, false, b1, (short)0, c11, false, false);
  }

  float s0 = 1.f, t0 = 0.f, s1 = 1.f, t1 = 0.f;
  if (g) {
    s0 = g[nc0] * BN_INV; t0 = beta[nc0];
    s1 = g[nc1] * BN_INV; t1 = beta[nc1];
  }
  const float bs0 = bias[nc0], bs1 = bias[nc1];
#pragma unroll
  for (int v = 0; v < 8; ++v) {
    const int ma = m0 + v + 8*h;        // C/D layout: M = v + 8*half, N = lane&15
    const int mb = ma + 16;
    if (ma < M) {
      Y[(size_t)ma * N + nc0] = apply_act((c00[v] + bs0) * s0 + t0, act);
      Y[(size_t)ma * N + nc1] = apply_act((c01[v] + bs1) * s1 + t1, act);
    }
    if (mb < M) {
      Y[(size_t)mb * N + nc0] = apply_act((c10[v] + bs0) * s0 + t0, act);
      Y[(size_t)mb * N + nc1] = apply_act((c11[v] + bs1) * s1 + t1, act);
    }
  }
}

// ============================================================================
// GEMM with on-the-fly f32 weights (used only for the 65536-K FF layer where
// W is consumed once and staging would double its HBM traffic).
// ============================================================================
__global__ void gemm_f32w(const float* __restrict__ X, const float* __restrict__ W,
                          const float* __restrict__ bias, const float* __restrict__ g,
                          const float* __restrict__ beta, float* __restrict__ Y,
                          int M, int K, int N, int act)
{
  const int lane = threadIdx.x & 31;
  const int wave = threadIdx.x >> 5;
  const int m0   = blockIdx.x * 32;
  const int n0   = blockIdx.y * 128 + wave * 32;
  if (n0 >= N) return;
  const int h  = lane >> 4;
  const int mn = lane & 15;

  const float* Xr0 = X + (size_t)(m0 + mn) * K;
  const float* Xr1 = X + (size_t)(m0 + 16 + mn) * K;
  const int nc0 = n0 + mn, nc1 = n0 + 16 + mn;

  f32x8 c00 = {0.f,0.f,0.f,0.f,0.f,0.f,0.f,0.f};
  f32x8 c01 = c00, c10 = c00, c11 = c00;

  for (int k0 = 0; k0 < K; k0 += 32) {
    bf16x16 a0, a1;
    load_a_frag(a0, Xr0, k0, h);
    load_a_frag(a1, Xr1, k0, h);
    bf16x16 b0, b1;
#pragma unroll
    for (int v = 0; v < 8; ++v) {
      const int kb = k0 + 2*v + 16*h;
      const float* wr0 = W + (size_t)kb * N;
      const float* wr1 = wr0 + N;
      b0[2*v]   = (__bf16)wr0[nc0];
      b0[2*v+1] = (__bf16)wr1[nc0];
      b1[2*v]   = (__bf16)wr0[nc1];
      b1[2*v+1] = (__bf16)wr1[nc1];
    }
    __builtin_prefetch(Xr0 + k0 + 32, 0, 3);
    __builtin_prefetch(W + (size_t)(k0 + 32) * N + nc0, 0, 3);

    c00 = __builtin_amdgcn_wmma_f32_16x16x32_bf16(false, a0, false, b0, (short)0, c00, false, false);
    c01 = __builtin_amdgcn_wmma_f32_16x16x32_bf16(false, a0, false, b1, (short)0, c01, false, false);
    c10 = __builtin_amdgcn_wmma_f32_16x16x32_bf16(false, a1, false, b0, (short)0, c10, false, false);
    c11 = __builtin_amdgcn_wmma_f32_16x16x32_bf16(false, a1, false, b1, (short)0, c11, false, false);
  }

  float s0 = 1.f, t0 = 0.f, s1 = 1.f, t1 = 0.f;
  if (g) {
    s0 = g[nc0] * BN_INV; t0 = beta[nc0];
    s1 = g[nc1] * BN_INV; t1 = beta[nc1];
  }
  const float bs0 = bias[nc0], bs1 = bias[nc1];
#pragma unroll
  for (int v = 0; v < 8; ++v) {
    const int ma = m0 + v + 8*h;
    const int mb = ma + 16;
    if (ma < M) {
      Y[(size_t)ma * N + nc0] = apply_act((c00[v] + bs0) * s0 + t0, act);
      Y[(size_t)ma * N + nc1] = apply_act((c01[v] + bs1) * s1 + t1, act);
    }
    if (mb < M) {
      Y[(size_t)mb * N + nc0] = apply_act((c10[v] + bs0) * s0 + t0, act);
      Y[(size_t)mb * N + nc1] = apply_act((c11[v] + bs1) * s1 + t1, act);
    }
  }
}

// ============================================================================
// Stage W[K,N] f32 -> bf16 WMMA B-fragment layout, zero-padded to Kpad.
// One thread per (kc, nt, lane): writes 16 contiguous bf16 (32 B).
// ============================================================================
__global__ void stage_weight_bf16(const float* __restrict__ W, int K, int N, int Kpad,
                                  __bf16* __restrict__ Wf)
{
  int gid = blockIdx.x * blockDim.x + threadIdx.x;
  int total = (Kpad >> 5) * (N >> 4) * 32;
  if (gid >= total) return;
  int lane = gid & 31;
  int t    = gid >> 5;
  int ntiles = N >> 4;
  int nt = t % ntiles;
  int kc = t / ntiles;
  int h  = lane >> 4;
  int mn = lane & 15;
  int n  = nt * 16 + mn;
  bf16x16 vals;
#pragma unroll
  for (int v = 0; v < 8; ++v) {
#pragma unroll
    for (int j = 0; j < 2; ++j) {
      int k = kc * 32 + 2*v + 16*h + j;
      float w = (k < K) ? W[(size_t)k * N + n] : 0.f;
      vals[2*v + j] = (__bf16)w;
    }
  }
  *reinterpret_cast<bf16x16*>(Wf + (size_t)gid * 16) = vals;
}

// ============================================================================
// Farthest point sampling: one block (256 threads) per batch.
// ============================================================================
__global__ void fps_kernel(const float* __restrict__ xyz, int Np, int npoint,
                           int* __restrict__ out_idx)
{
  const int b = blockIdx.x;
  const float* p = xyz + (size_t)b * Np * 3;
  __shared__ float dist[1024];
  __shared__ float rv[256];
  __shared__ int   ri[256];
  __shared__ int   cur;
  const int t = threadIdx.x;
  for (int i = t; i < Np; i += 256) dist[i] = 1e10f;
  if (t == 0) cur = 0;
  __syncthreads();
  for (int it = 0; it < npoint; ++it) {
    const int far = cur;
    if (t == 0) out_idx[b * npoint + it] = far;
    const float cx = p[far*3], cy = p[far*3+1], cz = p[far*3+2];
    float best = -1.f; int bi = 0;
    for (int i = t; i < Np; i += 256) {
      float dx = p[i*3] - cx, dy = p[i*3+1] - cy, dz = p[i*3+2] - cz;
      float d  = dx*dx + dy*dy + dz*dz;
      float dm = fminf(dist[i], d);
      dist[i] = dm;
      if (dm > best) { best = dm; bi = i; }
    }
    rv[t] = best; ri[t] = bi;
    __syncthreads();
    for (int s = 128; s > 0; s >>= 1) {
      if (t < s) {
        if (rv[t+s] > rv[t] || (rv[t+s] == rv[t] && ri[t+s] < ri[t])) {
          rv[t] = rv[t+s]; ri[t] = ri[t+s];
        }
      }
      __syncthreads();
    }
    if (t == 0) cur = ri[0];
    __syncthreads();
  }
}

__global__ void gather_xyz_kernel(const float* __restrict__ xyz, const int* __restrict__ idx,
                                  int Np, int S, float* __restrict__ out)
{
  int gid = blockIdx.x * blockDim.x + threadIdx.x;
  if (gid >= BT * S) return;
  int b = gid / S;
  int pi = idx[gid];
  out[gid*3+0] = xyz[((size_t)b*Np + pi)*3 + 0];
  out[gid*3+1] = xyz[((size_t)b*Np + pi)*3 + 1];
  out[gid*3+2] = xyz[((size_t)b*Np + pi)*3 + 2];
}

__global__ void ball_query_kernel(const float* __restrict__ xyz, const float* __restrict__ nxyz,
                                  int Np, int S, float r2, int nsample, int* __restrict__ idx)
{
  int gid = blockIdx.x * blockDim.x + threadIdx.x;
  if (gid >= BT * S) return;
  int b = gid / S;
  const float* p = xyz + (size_t)b * Np * 3;
  float cx = nxyz[gid*3], cy = nxyz[gid*3+1], cz = nxyz[gid*3+2];
  int* o = idx + (size_t)gid * nsample;
  int cnt = 0;
  for (int i = 0; i < Np && cnt < nsample; ++i) {
    float dx = p[i*3]-cx, dy = p[i*3+1]-cy, dz = p[i*3+2]-cz;
    if (dx*dx + dy*dy + dz*dz <= r2) o[cnt++] = i;
  }
  int fill = (cnt > 0) ? o[0] : 0;
  for (int k = cnt; k < nsample; ++k) o[k] = fill;
}

// grouped[b,j,s,:] = [xyz[idx]-new_xyz | feats[idx] | zeros up to Kpad]
__global__ void group_kernel(const float* __restrict__ xyz, const float* __restrict__ nxyz,
                             const float* __restrict__ feats, const int* __restrict__ idx,
                             int Np, int S, int nsample, int Cf, int Kpad,
                             float* __restrict__ out)
{
  int gid = blockIdx.x * blockDim.x + threadIdx.x;
  if (gid >= BT * S * nsample) return;
  int r = gid / nsample;            // b*S + j
  int b = r / S;
  int pi = idx[gid];
  size_t ob = (size_t)gid * Kpad;
  size_t pb = ((size_t)b * Np + pi) * 3;
  out[ob+0] = xyz[pb+0] - nxyz[r*3+0];
  out[ob+1] = xyz[pb+1] - nxyz[r*3+1];
  out[ob+2] = xyz[pb+2] - nxyz[r*3+2];
  if (Cf > 0) {
    const float* f = feats + ((size_t)b * Np + pi) * Cf;
    for (int c = 0; c < Cf; ++c) out[ob + 3 + c] = f[c];
  }
  for (int c = 3 + Cf; c < Kpad; ++c) out[ob + c] = 0.f;
}

__global__ void maxpool_kernel(const float* __restrict__ in, int S, int nsample, int C,
                               float* __restrict__ out)
{
  int gid = blockIdx.x * blockDim.x + threadIdx.x;
  if (gid >= BT * S * C) return;
  int r = gid / C, c = gid % C;
  float m = -1e30f;
  for (int s = 0; s < nsample; ++s)
    m = fmaxf(m, in[((size_t)r * nsample + s) * C + c]);
  out[gid] = m;
}

__global__ void knn3_kernel(const float* __restrict__ xyz1, const float* __restrict__ xyz2,
                            int N1, int N2, int* __restrict__ oi, float* __restrict__ ow)
{
  int gid = blockIdx.x * blockDim.x + threadIdx.x;
  if (gid >= BT * N1) return;
  int b = gid / N1;
  float cx = xyz1[gid*3], cy = xyz1[gid*3+1], cz = xyz1[gid*3+2];
  const float* q = xyz2 + (size_t)b * N2 * 3;
  float bd0 = 1e30f, bd1 = 1e30f, bd2 = 1e30f;
  int   bi0 = 0,     bi1 = 0,     bi2 = 0;
  for (int i = 0; i < N2; ++i) {
    float dx = q[i*3]-cx, dy = q[i*3+1]-cy, dz = q[i*3+2]-cz;
    float d = dx*dx + dy*dy + dz*dz;
    if (d < bd0)      { bd2=bd1; bi2=bi1; bd1=bd0; bi1=bi0; bd0=d; bi0=i; }
    else if (d < bd1) { bd2=bd1; bi2=bi1; bd1=d; bi1=i; }
    else if (d < bd2) { bd2=d; bi2=i; }
  }
  float w0 = 1.f/(bd0+1e-8f), w1 = 1.f/(bd1+1e-8f), w2 = 1.f/(bd2+1e-8f);
  float inv = 1.f/(w0+w1+w2);
  oi[gid*3+0] = bi0; oi[gid*3+1] = bi1; oi[gid*3+2] = bi2;
  ow[gid*3+0] = w0*inv; ow[gid*3+1] = w1*inv; ow[gid*3+2] = w2*inv;
}

// out[b,n,:] = [p1[b,n,:] | sum_k w_k * p2[b,idx_k,:]]
__global__ void interp_concat_kernel(const float* __restrict__ p1, const float* __restrict__ p2,
                                     const int* __restrict__ ki, const float* __restrict__ kw,
                                     int N1, int N2, int C1, int C2, float* __restrict__ out)
{
  size_t gid = (size_t)blockIdx.x * blockDim.x + threadIdx.x;
  int C = C1 + C2;
  if (gid >= (size_t)BT * N1 * C) return;
  size_t r = gid / C;               // b*N1 + n
  int   c = (int)(gid % C);
  int   b = (int)(r / N1);
  if (c < C1) {
    out[gid] = p1[r * C1 + c];
  } else {
    int c2 = c - C1;
    float acc = 0.f;
    for (int k = 0; k < 3; ++k)
      acc += kw[r*3+k] * p2[((size_t)b * N2 + ki[r*3+k]) * C2 + c2];
    out[gid] = acc;
  }
}

// (BT,1024,64) -> (BT, 64*1024) with layout [bt][c*1024+n]
__global__ void transpose_cn_kernel(const float* __restrict__ in, float* __restrict__ out)
{
  int gid = blockIdx.x * blockDim.x + threadIdx.x;
  if (gid >= BT * 64 * 1024) return;
  int bt  = gid / (64 * 1024);
  int rem = gid % (64 * 1024);
  int c = rem / 1024, n = rem % 1024;
  out[gid] = in[((size_t)bt * 1024 + n) * 64 + c];
}

// ============================================================================
// Host driver
// ============================================================================
struct Layer { const float *W, *b, *g, *beta; int cin, cout; const __bf16* Wf; int kpad; };

extern "C" void kernel_launch(void* const* d_in, const int* in_sizes, int n_in,
                              void* d_out, int out_size, void* d_ws, size_t ws_size,
                              hipStream_t stream)
{
  (void)out_size; (void)ws_size;
  // ---- locate 'points' by unique flat size; rest are params in JAX
  //      tree-flatten (sorted-key) order: conv1, ff, fp1..fp4, sa1..sa4 ----
  int pidx = -1;
  for (int i = 0; i < n_in; ++i) if (in_sizes[i] == 8*2*1024*3) { pidx = i; break; }
  const float* points = (const float*)d_in[pidx];
  const float* P[128]; int npar = 0;
  for (int i = 0; i < n_in && npar < 128; ++i) if (i != pidx) P[npar++] = (const float*)d_in[i];

  int cu = 0;
  auto take4 = [&](int ci, int co){ Layer L{P[cu],P[cu+1],P[cu+2],P[cu+3],ci,co,nullptr,ci}; cu += 4; return L; };
  auto take2 = [&](int ci, int co){ Layer L{P[cu],P[cu+1],nullptr,nullptr,ci,co,nullptr,ci}; cu += 2; return L; };

  Layer conv1  = take4(128, 64);
  Layer ffL[3] = { take4(65536, 1024), take4(1024, 512), take2(512, 128) };
  Layer fp1[3] = { take4(128,128), take4(128,128), take4(128,128) };
  Layer fp2[2] = { take4(320,256), take4(256,128) };
  Layer fp3[2] = { take4(384,256), take4(256,256) };
  Layer fp4[2] = { take4(768,256), take4(256,256) };
  Layer sa1[3] = { take4(3,32),    take4(32,32),   take4(32,64)   };
  Layer sa2[3] = { take4(67,64),   take4(64,64),   take4(64,128)  };
  Layer sa3[3] = { take4(131,128), take4(128,128), take4(128,256) };
  Layer sa4[3] = { take4(259,256), take4(256,256), take4(256,512) };

  // ---- workspace carve (floats; all block sizes are multiples of 4) ----
  float* wbase = (float*)d_ws;
  size_t off = 0;
  auto alloc = [&](size_t n){ float* p = wbase + off; off += n; return p; };
  int*   fps_idx  = (int*)alloc(16*512);
  float* l1x = alloc(16*512*3);
  float* l2x = alloc(16*256*3);
  float* l3x = alloc(16*64*3);
  float* l4x = alloc(16*16*3);
  int*   ball_idx = (int*)alloc((size_t)16*512*32);
  float* bufin = alloc(12582912);    // max grouped row-padded (sa2: 131072*96) / concat
  float* actA  = alloc(16777216);    // 262144*64
  float* actB  = alloc(16777216);
  float* l1f = alloc(16*512*64);
  float* l2f = alloc(16*256*128);
  float* l3f = alloc(16*64*256);
  float* l4f = alloc(16*16*512);
  float* l3p = alloc(16*64*256);
  float* l2p = alloc(16*256*256);
  float* l1p = alloc(16*512*128);
  float* l0  = alloc(16*1024*128);
  float* cvo = alloc(16*1024*64);
  float* xt  = alloc(32*65536);      // 32 rows: rows 16..31 are padding
  float* f1  = alloc(32*1024);
  float* f2  = alloc(32*512);
  int*   kni = (int*)alloc(16*1024*3);
  float* knw = alloc(16*1024*3);

  // ---- stage weights into bf16 fragment layout (all but the 65536-K ff1) ----
  auto stageL = [&](Layer& L){
    int Kpad = ((L.cin + 31) / 32) * 32;
    __bf16* dst = (__bf16*)alloc(((size_t)Kpad * L.cout) / 2);
    int tot = (Kpad/32) * (L.cout/16) * 32;
    stage_weight_bf16<<<(tot+255)/256, 256, 0, stream>>>(L.W, L.cin, L.cout, Kpad, dst);
    L.Wf = dst; L.kpad = Kpad;
  };
  stageL(conv1);
  stageL(ffL[1]); stageL(ffL[2]);
  for (int i = 0; i < 3; ++i) stageL(fp1[i]);
  for (int i = 0; i < 2; ++i) { stageL(fp2[i]); stageL(fp3[i]); stageL(fp4[i]); }
  for (int i = 0; i < 3; ++i) { stageL(sa1[i]); stageL(sa2[i]); stageL(sa3[i]); stageL(sa4[i]); }

  auto gemm = [&](const float* X, const Layer& L, float* Y, int M, int act){
    dim3 grid((unsigned)((M + 31)/32), (unsigned)((L.cout + 127)/128));
    if (L.Wf)
      gemm_fw<<<grid, dim3(128,1,1), 0, stream>>>(X, L.Wf, L.b, L.g, L.beta, Y,
                                                  M, L.kpad, L.cout, act);
    else
      gemm_f32w<<<grid, dim3(128,1,1), 0, stream>>>(X, L.W, L.b, L.g, L.beta, Y,
                                                    M, L.cin, L.cout, act);
  };

  auto set_abs = [&](const float* xyz_in, int Np, const float* feats, int Cf,
                     int S, float radius, const Layer* Ls, float* nxyz, float* outf){
    fps_kernel<<<16, 256, 0, stream>>>(xyz_in, Np, S, fps_idx);
    int n1 = 16*S;
    gather_xyz_kernel<<<(n1+255)/256, 256, 0, stream>>>(xyz_in, fps_idx, Np, S, nxyz);
    ball_query_kernel<<<(n1+255)/256, 256, 0, stream>>>(xyz_in, nxyz, Np, S,
                                                        radius*radius, 32, ball_idx);
    int n2 = 16*S*32;
    int Kpad = Ls[0].kpad;           // padded to multiple of 32
    group_kernel<<<(n2+255)/256, 256, 0, stream>>>(xyz_in, nxyz, feats, ball_idx,
                                                   Np, S, 32, Cf, Kpad, bufin);
    gemm(bufin, Ls[0], actA, n2, 1);
    gemm(actA,  Ls[1], actB, n2, 1);
    gemm(actB,  Ls[2], actA, n2, 1);
    int n3 = 16*S*Ls[2].cout;
    maxpool_kernel<<<(n3+255)/256, 256, 0, stream>>>(actA, S, 32, Ls[2].cout, outf);
  };

  auto fprop = [&](const float* xyz1, int N1, const float* xyz2, int N2,
                   const float* p1, int C1, const float* p2, int C2,
                   const Layer* Ls, int nL, float* outp){
    knn3_kernel<<<(16*N1+255)/256, 256, 0, stream>>>(xyz1, xyz2, N1, N2, kni, knw);
    size_t tot = (size_t)16 * N1 * (C1 + C2);
    interp_concat_kernel<<<(unsigned)((tot+255)/256), 256, 0, stream>>>(
        p1, p2, kni, knw, N1, N2, C1, C2, bufin);
    int M = 16*N1;
    const float* X = bufin;
    for (int i = 0; i < nL; ++i) {
      float* Y = (i == nL-1) ? outp : ((i & 1) ? actB : actA);
      gemm(X, Ls[i], Y, M, 1);
      X = Y;
    }
  };

  // ---- encoder ----
  set_abs(points, 1024, nullptr, 0,   512, 0.1f, sa1, l1x, l1f);
  set_abs(l1x,    512,  l1f,    64,   256, 0.2f, sa2, l2x, l2f);
  set_abs(l2x,    256,  l2f,    128,  64,  0.4f, sa3, l3x, l3f);
  set_abs(l3x,    64,   l3f,    256,  16,  0.8f, sa4, l4x, l4f);

  fprop(l3x,    64,   l4x, 16,  l3f,     256, l4f, 512, fp4, 2, l3p);
  fprop(l2x,    256,  l3x, 64,  l2f,     128, l3p, 256, fp3, 2, l2p);
  fprop(l1x,    512,  l2x, 256, l1f,     64,  l2p, 256, fp2, 2, l1p);
  fprop(points, 1024, l1x, 512, nullptr, 0,   l1p, 128, fp1, 3, l0);

  // ---- head ----
  gemm(l0, conv1, cvo, 16*1024, 1);
  transpose_cn_kernel<<<(16*64*1024+255)/256, 256, 0, stream>>>(cvo, xt);
  gemm(xt, ffL[0], f1, 16, 2);
  gemm(f1, ffL[1], f2, 16, 2);
  gemm(f2, ffL[2], (float*)d_out, 16, 0);
}
